// GATlayer_47141561041410
// MI455X (gfx1250) — matrix operand order repaired and checked by reading, not compiled, not run on hardware
//
#include <hip/hip_runtime.h>
#include <hip/hip_bf16.h>

// GAT layer, fused for MI455X (gfx1250, wave32, WMMA).
// deg+bond (134MB) fit in 192MB L2 -> recompute mask(i,j) on the fly; never
// materialize mask or the [H,N,N] attention matrix. Softmax is over axis=1
// (columns): per-(h,j) normalizer lse[h,j]=max+logsumexp computed online.
// Attention GEMM: P=exp(score-lse) staged per block into a double-buffered LDS
// tile (bf16, WMMA A-fragment order, conflict-free pad), proj pre-swizzled in
// memory into WMMA B-fragment order -> inner loop = ds_load_b128 +
// global_load_b128 + v_wmma, one barrier per 64-j step, global_prefetch ahead.

#define N_NODES 4096
#define F_IN    128
#define F_OUTC  64
#define NH      4
#define NEG_INF -1e9f
#define LN_EPS  1e-5f
#define NJB     (N_NODES / 32)     // 128 j-blocks of 32
#define JSTEP   64                 // j columns per staged step (2 WMMAs)
#define PROW    72                 // LDS P row stride in bf16 (144B: 36m mod 64
                                   // -> 16 distinct banks, 16B-aligned runs)

typedef __attribute__((ext_vector_type(16))) __bf16         v16bf;
typedef __attribute__((ext_vector_type(8)))  float          v8f;
typedef __attribute__((ext_vector_type(8)))  unsigned short u16x8;

union BfFrag { v16bf v; unsigned short u[16]; u16x8 q[2]; };

// float -> bf16 bits, round-to-nearest-even (pure integer ops)
__device__ __forceinline__ unsigned short f2bf(float x) {
    unsigned int b = __float_as_uint(x);
    b += 0x7FFFu + ((b >> 16) & 1u);
    return (unsigned short)(b >> 16);
}

__device__ __forceinline__ float mask_val(float deg, float bond, float cutoff) {
    float wdm = deg + bond;
    return (wdm > 0.f) ? wdm : ((bond > cutoff) ? (bond + wdm) : NEG_INF);
}

// ---------------------------------------------------------------------------
// Generic 16x16 wave tile: C = A(16xK) * B(Kx16), bf16 WMMA, K multiple of 32.
// A element (m,k) at A[m*lda + k]; B element (k,n) at B[k*ldb_k + n*ldb_n].
// Lane layouts per CDNA5 ISA 7.12.2.
// ---------------------------------------------------------------------------
__device__ __forceinline__ v8f wmma_tile_bf16(const float* __restrict__ A, int lda,
                                              const float* __restrict__ B,
                                              int ldb_k, int ldb_n, int K) {
    int lane = threadIdx.x & 31;
    int mrow = lane & 15;
    int hi   = lane >> 4;
    v8f c = {};
    for (int k0 = 0; k0 < K; k0 += 32) {
        BfFrag a, b;
#pragma unroll
        for (int e = 0; e < 16; ++e) {
            int ka = (e < 8) ? (e + (hi ? 8 : 0)) : (e - 8 + (hi ? 24 : 16));
            a.u[e] = f2bf(A[(size_t)mrow * lda + k0 + ka]);
            int kb = ka;
            b.u[e] = f2bf(B[(size_t)(k0 + kb) * ldb_k + (size_t)mrow * ldb_n]);
        }
        c = __builtin_amdgcn_wmma_f32_16x16x32_bf16(false, a.v, false, b.v,
                                                    (short)0, c, false, false);
    }
    return c;
}

// ---------------------------------------------------------------------------
// Kernel 1: mask + fresh LayerNorm over last dim -> mask_ln output.
// ---------------------------------------------------------------------------
__global__ void mask_ln_kernel(const float* __restrict__ deg,
                               const float* __restrict__ bond,
                               const int* __restrict__ cut_i,
                               float* __restrict__ mask_ln) {
    __shared__ float row[N_NODES];
    __shared__ float sbuf[256], ssbuf[256];
    float cutoff = (float)(*cut_i);
    int i = blockIdx.x, t = threadIdx.x;
    const float* dr = deg  + (size_t)i * N_NODES;
    const float* br = bond + (size_t)i * N_NODES;
    float s = 0.f, ss = 0.f;
    for (int j = t; j < N_NODES; j += 256) {
        float m = mask_val(dr[j], br[j], cutoff);
        row[j] = m;
        s += m; ss += m * m;
    }
    sbuf[t] = s; ssbuf[t] = ss;
    __syncthreads();
    for (int o = 128; o > 0; o >>= 1) {
        if (t < o) { sbuf[t] += sbuf[t + o]; ssbuf[t] += ssbuf[t + o]; }
        __syncthreads();
    }
    float mu   = sbuf[0] * (1.f / N_NODES);
    float var  = ssbuf[0] * (1.f / N_NODES) - mu * mu;
    float rstd = rsqrtf(var + LN_EPS);
    for (int j = t; j < N_NODES; j += 256)
        mask_ln[(size_t)i * N_NODES + j] = (row[j] - mu) * rstd;
}

// ---------------------------------------------------------------------------
// Kernel 2: proj[h][n][f] = X @ Wp[h] (WMMA). Stores f32 (for score dots) and
// a bf16 copy pre-swizzled into WMMA B-fragment order for the attention GEMM:
//   frag[h][jb][ftile][lane][e], j = jb*32 + (e + (lane>>4)*16), f = ftile*16 + (lane&15)
// ---------------------------------------------------------------------------
__global__ void proj_kernel(const float* __restrict__ X,
                            const float* __restrict__ Wp,      // [H][F_IN][F_OUT]
                            float* __restrict__ proj,          // [H][N][F_OUT]
                            unsigned short* __restrict__ projf) {
    int w = threadIdx.x >> 5, lane = threadIdx.x & 31;
    int h  = blockIdx.y;
    int i0 = blockIdx.x * 32 + (w & 1) * 16;
    int f0 = (w >> 1) * 16;
    const float* A = X + (size_t)i0 * F_IN;
    const float* B = Wp + (size_t)h * F_IN * F_OUTC + f0;   // (k,n) at k*F_OUT + n
    v8f c = wmma_tile_bf16(A, F_IN, B, F_OUTC, 1, F_IN);
    int nn = lane & 15, hi = lane >> 4;
#pragma unroll
    for (int r = 0; r < 8; ++r) {
        int j = i0 + r + (hi ? 8 : 0);                 // proj row (attention j)
        int f = f0 + nn;
        proj[((size_t)h * N_NODES + j) * F_OUTC + f] = c[r];
        // B-fragment swizzle
        int jb = j >> 5, kk = j & 31;
        int hia = kk >> 4, e = kk & 15;
        int lane_a = hia * 16 + (f & 15);
        int ftile  = (f >> 4) & 3;
        size_t fo = ((((size_t)h * NJB + jb) * 4 + ftile) * 32 + lane_a) * 16 + e;
        projf[fo] = f2bf(c[r]);
    }
}

// ---------------------------------------------------------------------------
// Kernel 3: skip[n][c] = X @ skip_w.T (WMMA). grid (N/32, 4), block 256.
// ---------------------------------------------------------------------------
__global__ void skip_kernel(const float* __restrict__ X,
                            const float* __restrict__ Wsk,     // [256][128]
                            float* __restrict__ skip) {        // [N][256]
    int w = threadIdx.x >> 5, lane = threadIdx.x & 31;
    int i0 = blockIdx.x * 32 + (w & 1) * 16;
    int f0 = blockIdx.y * 64 + (w >> 1) * 16;
    const float* A = X + (size_t)i0 * F_IN;
    const float* B = Wsk + (size_t)f0 * F_IN;     // (k,n) at n*F_IN + k (transposed)
    v8f c = wmma_tile_bf16(A, F_IN, B, 1, F_IN, F_IN);
    int nn = lane & 15, hi = lane >> 4;
#pragma unroll
    for (int r = 0; r < 8; ++r) {
        int mi = r + (hi ? 8 : 0);
        skip[(size_t)(i0 + mi) * (NH * F_OUTC) + f0 + nn] = c[r];
    }
}

// ---------------------------------------------------------------------------
// Kernel 4: s_src/s_tgt scalar scores per (h,node): length-64 dots.
// ---------------------------------------------------------------------------
__global__ void dots_kernel(const float* __restrict__ proj,
                            const float* __restrict__ wsrc,    // [H][F_OUT][1]
                            const float* __restrict__ wtgt,
                            float* __restrict__ ssrc, float* __restrict__ stgt) {
    int idx = blockIdx.x * blockDim.x + threadIdx.x;           // = h*N + n
    int h = idx >> 12;
    const float* p = proj + (size_t)idx * F_OUTC;
    const float* a = wsrc + h * F_OUTC;
    const float* b = wtgt + h * F_OUTC;
    float sa = 0.f, sb = 0.f;
#pragma unroll 8
    for (int f = 0; f < F_OUTC; ++f) { float v = p[f]; sa += v * a[f]; sb += v * b[f]; }
    ssrc[idx] = sa; stgt[idx] = sb;
}

// ---------------------------------------------------------------------------
// Kernel 5: online column logsumexp over i for each (h,j); packs the per-column
// params the attention kernel needs into one float2: (s_tgt, max + logsumexp).
// Block: 256 threads = 4 heads x 64 columns; ssrc staged in LDS per chunk;
// global_prefetch the upcoming deg/bond rows.
// ---------------------------------------------------------------------------
__global__ void colstat_kernel(const float* __restrict__ deg,
                               const float* __restrict__ bond,
                               const int* __restrict__ cut_i,
                               const float* __restrict__ ssrc,
                               const float* __restrict__ stgt,
                               float* __restrict__ colp) {     // float2 per (h,j)
    __shared__ float sh[NH * 2048];                            // 32 KB
    float cutoff = (float)(*cut_i);
    int t  = threadIdx.x;
    int h  = t >> 6;
    int j  = blockIdx.x * 64 + (t & 63);
    float st = stgt[h * N_NODES + j];
    float m = -3.0e38f, Z = 0.f;
    for (int i0 = 0; i0 < N_NODES; i0 += 2048) {
        for (int k = t; k < NH * 2048; k += 256)
            sh[k] = ssrc[(k >> 11) * N_NODES + i0 + (k & 2047)];
        __syncthreads();
        for (int ii = 0; ii < 2048; ++ii) {
            int i = i0 + ii;
            if (i + 8 < N_NODES) {   // pull upcoming rows toward the WGP
                __builtin_prefetch(&deg[(size_t)(i + 8) * N_NODES + j], 0, 1);
                __builtin_prefetch(&bond[(size_t)(i + 8) * N_NODES + j], 0, 1);
            }
            float msk = mask_val(deg[(size_t)i * N_NODES + j],
                                 bond[(size_t)i * N_NODES + j], cutoff);
            float x = sh[h * 2048 + ii] + st;
            x = (x > 0.f) ? x : 0.2f * x;
            x += msk;
            if (x > m) { Z = Z * __expf(m - x) + 1.f; m = x; }
            else        Z += __expf(x - m);
        }
        __syncthreads();
    }
    colp[2 * (h * N_NODES + j) + 0] = st;
    colp[2 * (h * N_NODES + j) + 1] = m + __logf(Z);
}

// ---------------------------------------------------------------------------
// Kernel 6: fused attention GEMM + skip + ELU.
// Block = 512 threads = 16 waves; i-tile = 16 rows, all 4 heads x 4 f-tiles.
// Double-buffered LDS P tile, JSTEP=64 columns per stage, ONE barrier per step:
//   stage: wave w owns P-row i0+w (lane=k, coalesced deg/bond loads), computes
//          exp(score-lse) for all heads -> bf16 LDS (A-fragment row order).
//   compute: per 32-j half: A = two ds_load_b128, B = two contiguous
//          global_load_b128 from pre-swizzled projf, one v_wmma.
// Overwrite of buffer (t mod 2) at step t+2 is safe: every wave's step-t LDS
// reads complete before it signals the step t+1 barrier.
// ---------------------------------------------------------------------------
__global__ void attn_kernel(const float* __restrict__ deg,
                            const float* __restrict__ bond,
                            const int* __restrict__ cut_i,
                            const unsigned short* __restrict__ projf,
                            const float* __restrict__ ssrc,
                            const float* __restrict__ colp,
                            const float* __restrict__ skip,
                            float* __restrict__ out) {
    __shared__ unsigned short smP[2][NH * 16 * PROW];          // 2 x 9 KB
    float cutoff = (float)(*cut_i);
    int w = threadIdx.x >> 5, lane = threadIdx.x & 31;
    int h  = w >> 2;
    int ft = w & 3;
    int i0 = blockIdx.x * 16;
    int mrow = lane & 15, hi = lane >> 4;

    float srow[NH];
#pragma unroll
    for (int hh = 0; hh < NH; ++hh) srow[hh] = ssrc[hh * N_NODES + i0 + w];

    const float*  dr = deg  + (size_t)(i0 + w) * N_NODES;
    const float*  br = bond + (size_t)(i0 + w) * N_NODES;
    const float2* cp = (const float2*)colp;

    v8f c = {};
    int buf = 0;
    for (int j0 = 0; j0 < N_NODES; j0 += JSTEP) {
        // prefetch next step's deg/bond lines (global_prefetch_b8)
        if (j0 + JSTEP < N_NODES) {
            __builtin_prefetch(dr + j0 + JSTEP + lane, 0, 1);
            __builtin_prefetch(br + j0 + JSTEP + lane, 0, 1);
        }
        // ---- stage P tile (one row per wave, two 32-column chunks)
        unsigned short* sp = smP[buf];
#pragma unroll
        for (int half = 0; half < 2; ++half) {
            int k = lane + half * 32;
            float msk = mask_val(dr[j0 + k], br[j0 + k], cutoff);
#pragma unroll
            for (int hh = 0; hh < NH; ++hh) {
                float2 sc = cp[hh * N_NODES + j0 + k];
                float x = srow[hh] + sc.x;
                x = (x > 0.f) ? x : 0.2f * x;
                sp[(hh * 16 + w) * PROW + k] = f2bf(__expf(x + msk - sc.y));
            }
        }
        __syncthreads();
        // ---- fragments + 2 WMMAs
        const unsigned short* rp = &smP[buf][(h * 16 + mrow) * PROW];
#pragma unroll
        for (int half = 0; half < 2; ++half) {
            BfFrag a, b;
            a.q[0] = *(const u16x8*)&rp[half * 32 + hi * 16];
            a.q[1] = *(const u16x8*)&rp[half * 32 + hi * 16 + 8];
            const u16x8* bp = (const u16x8*)(projf +
                ((((size_t)h * NJB + (j0 >> 5) + half) * 4 + ft) * 32 + lane) * 16);
            b.q[0] = bp[0];
            b.q[1] = bp[1];
            c = __builtin_amdgcn_wmma_f32_16x16x32_bf16(false, a.v, false, b.v,
                                                        (short)0, c, false, false);
        }
        buf ^= 1;
    }
    // ---- epilogue: + skip, ELU
#pragma unroll
    for (int r = 0; r < 8; ++r) {
        int row = i0 + r + (hi ? 8 : 0);
        int col = h * F_OUTC + ft * 16 + mrow;
        float v = c[r] + skip[(size_t)row * (NH * F_OUTC) + col];
        v = (v > 0.f) ? v : (__expf(v) - 1.f);                 // ELU
        out[(size_t)row * (NH * F_OUTC) + col] = v;
    }
}

// ---------------------------------------------------------------------------
extern "C" void kernel_launch(void* const* d_in, const int* in_sizes, int n_in,
                              void* d_out, int out_size, void* d_ws, size_t ws_size,
                              hipStream_t stream) {
    const float* X    = (const float*)d_in[0];   // [4096][128]
    const float* deg  = (const float*)d_in[1];   // [4096][4096]
    // d_in[2] edges_features_distance: unused by the reference output
    const float* bond = (const float*)d_in[3];   // [4096][4096]
    const float* Wp   = (const float*)d_in[4];   // [4][128][64]
    const float* wsrc = (const float*)d_in[5];   // [4][64][1]
    const float* wtgt = (const float*)d_in[6];   // [4][64][1]
    const float* Wsk  = (const float*)d_in[7];   // [256][128]
    const int*   cut  = (const int*)d_in[8];     // scalar

    float* out     = (float*)d_out;                                // [4096][256]
    float* mask_ln = out + (size_t)N_NODES * NH * F_OUTC;          // [4096][4096]

    // Workspace layout (~10.7 MB total)
    char* ws = (char*)d_ws;
    float*          proj  = (float*)(ws);                    // 4 MB   f32 proj
    unsigned short* projf = (unsigned short*)(ws + 4194304); // 2 MB   bf16 B-fragments
    float*          ssrc  = (float*)(ws + 6291456);          // 64 KB
    float*          stgt  = (float*)(ws + 6356992);          // 64 KB
    float*          colp  = (float*)(ws + 6422528);          // 128 KB (stgt,lse) pairs
    float*          skip  = (float*)(ws + 6553600);          // 4 MB

    mask_ln_kernel<<<N_NODES, 256, 0, stream>>>(deg, bond, cut, mask_ln);
    proj_kernel<<<dim3(N_NODES / 32, NH), 256, 0, stream>>>(X, Wp, proj, projf);
    skip_kernel<<<dim3(N_NODES / 32, (NH * F_OUTC) / 64), 256, 0, stream>>>(X, Wsk, skip);
    dots_kernel<<<(NH * N_NODES) / 256, 256, 0, stream>>>(proj, wsrc, wtgt, ssrc, stgt);
    colstat_kernel<<<N_NODES / 64, 256, 0, stream>>>(deg, bond, cut, ssrc, stgt, colp);
    attn_kernel<<<N_NODES / 16, 512, 0, stream>>>(deg, bond, cut, projf, ssrc,
                                                  colp, skip, out);
}
